// GATSelfAttention_50869592654344
// MI455X (gfx1250) — matrix operand order repaired and checked by reading, not compiled, not run on hardware
//
#include <hip/hip_runtime.h>
#include <hip/hip_bf16.h>
#include <stdint.h>

// ---------------------------------------------------------------------------
// GAT self-attention for MI455X (gfx1250, wave32, WMMA).
// Sizes: N=64, E=512, D=512, T=3.
// Pipeline:
//   k_f32_to_bf16 : X -> bf16, W_type[2] -> bf16
//   k_gates       : g = sigmoid(relu(q@W1)@W2)             [3,64,1024]
//   k_uvec        : u[t,s,n,:] = W_t @ (g_s * a_s)         [3,2,64,512]
//   k_leftright   : lr[t,s,n,e] = X[n,e,:] . u[t,s,n,:]    [3,2,64,512]
//   k_gemm (bf16) : h2 = X @ W2            (v_wmma_f32_16x16x32_bf16)
//   k_coefT       : softmax(scores) written TRANSPOSED as bf16 [n,j,i]
//   k_gemm (f32)  : out = coefsT @ h2      (v_wmma_f32_16x16x32_bf16)
// ---------------------------------------------------------------------------

#define NEG_INF_F   (-9e15f)
#define LRELU_SLOPE 0.2f

typedef __attribute__((ext_vector_type(16))) __bf16 v16bf;
typedef __attribute__((ext_vector_type(8)))  float  v8f;
typedef __attribute__((ext_vector_type(4)))  int    v4i;

static __device__ __forceinline__ unsigned short f2bf(float f) {
  unsigned int u = __builtin_bit_cast(unsigned int, f);
  unsigned int r = u + 0x7FFFu + ((u >> 16) & 1u);   // round-to-nearest-even
  return (unsigned short)(r >> 16);
}

// ---------------------------------------------------------------------------
__global__ void k_f32_to_bf16(const float* __restrict__ src,
                              unsigned short* __restrict__ dst, long n) {
  long i = (long)blockIdx.x * blockDim.x + threadIdx.x;
  long stride = (long)gridDim.x * blockDim.x;
  for (; i < n; i += stride) dst[i] = f2bf(src[i]);
}

// --------------------------------------------------------------------------- gates
__global__ void k_gates(const float* __restrict__ query,  // [64,512]
                        const float* __restrict__ W1,     // [3,512,1024]
                        const float* __restrict__ W2,     // [3,1024,1024]
                        float* __restrict__ gates)        // [3,64,1024]
{
  const int n = blockIdx.x, t = blockIdx.y, tid = threadIdx.x; // 256 thr
  __shared__ float qs[512];
  __shared__ float g1[1024];
  for (int k = tid; k < 512; k += 256) qs[k] = query[n * 512 + k];
  __syncthreads();
  const float* w1 = W1 + (long)t * 512 * 1024;
  for (int it = 0; it < 4; ++it) {
    int j = tid + it * 256;
    float acc = 0.f;
    for (int k = 0; k < 512; ++k) acc = fmaf(qs[k], w1[(long)k * 1024 + j], acc);
    g1[j] = acc > 0.f ? acc : 0.f;
  }
  __syncthreads();
  const float* w2 = W2 + (long)t * 1024 * 1024;
  for (int it = 0; it < 4; ++it) {
    int j = tid + it * 256;
    float acc = 0.f;
    for (int k = 0; k < 1024; ++k) acc = fmaf(g1[k], w2[(long)k * 1024 + j], acc);
    gates[((long)t * 64 + n) * 1024 + j] = 1.f / (1.f + __expf(-acc));
  }
}

// --------------------------------------------------------------------------- folded u = W_t @ (g*a)
__global__ void k_uvec(const float* __restrict__ W_type,  // [3,512,512]
                       const float* __restrict__ a_type,  // [3,1024]
                       const float* __restrict__ gates,   // [3,64,1024]
                       float* __restrict__ u)             // [3,2,64,512]
{
  const int n = blockIdx.x, t = blockIdx.y, tid = threadIdx.x; // 256 thr
  __shared__ float va[2][512];
  for (int idx = tid; idx < 1024; idx += 256) {
    int s = idx >> 9, dd = idx & 511;
    va[s][dd] = gates[((long)t * 64 + n) * 1024 + idx] * a_type[t * 1024 + idx];
  }
  __syncthreads();
  const float* w = W_type + (long)t * 512 * 512;
  for (int kk = 0; kk < 2; ++kk) {
    int k = tid + kk * 256;
    const float* wrow = w + (long)k * 512;
    float a0 = 0.f, a1 = 0.f;
    for (int d = 0; d < 512; ++d) {
      float wv = wrow[d];
      a0 = fmaf(wv, va[0][d], a0);
      a1 = fmaf(wv, va[1][d], a1);
    }
    u[(((long)t * 2 + 0) * 64 + n) * 512 + k] = a0;
    u[(((long)t * 2 + 1) * 64 + n) * 512 + k] = a1;
  }
}

// --------------------------------------------------------------------------- lr[q,n,e] = X[n,e,:].u[q,n,:]
__global__ void k_leftright(const float* __restrict__ x,  // [64,512,512]
                            const float* __restrict__ u,  // [6,64,512]
                            float* __restrict__ lr)       // [6,64,512]
{
  const int n = blockIdx.x, etile = blockIdx.y, tid = threadIdx.x; // 256 thr
  const int r = tid >> 4, s = tid & 15;
  __shared__ float us[6][512];
  __shared__ float red[16][6][16];
  for (int idx = tid; idx < 6 * 512; idx += 256) {
    int q = idx >> 9, k = idx & 511;
    us[q][k] = u[((long)q * 64 + n) * 512 + k];
  }
  __syncthreads();
  int e = etile * 16 + r;
  const float* xrow = x + ((long)n * 512 + e) * 512;
  float p[6] = {0.f, 0.f, 0.f, 0.f, 0.f, 0.f};
  for (int k = s; k < 512; k += 16) {
    float xv = xrow[k];
#pragma unroll
    for (int q = 0; q < 6; ++q) p[q] = fmaf(xv, us[q][k], p[q]);
  }
#pragma unroll
  for (int q = 0; q < 6; ++q) red[r][q][s] = p[q];
  __syncthreads();
  if (s < 6) {
    float acc = 0.f;
#pragma unroll
    for (int i = 0; i < 16; ++i) acc += red[r][s][i];
    lr[((long)s * 64 + n) * 512 + e] = acc;
  }
}

// --------------------------------------------------------------------------- scores -> softmax -> coefsT (bf16)
__global__ void k_coefT(const int* __restrict__ adj,        // [64,512,512]
                        const float* __restrict__ lr,       // [6,64,512]
                        unsigned short* __restrict__ coefT) // bf16 [64,512,512] (j-major)
{
  const int i = blockIdx.x, n = blockIdx.y, tid = threadIdx.x; // 128 thr
  __shared__ float rv[3][512];
  __shared__ float red[128];
  __shared__ float sc[512];
  float lv[3];
#pragma unroll
  for (int t = 0; t < 3; ++t) lv[t] = lr[(((long)t * 2 + 0) * 64 + n) * 512 + i];
  for (int idx = tid; idx < 3 * 512; idx += 128) {
    int t = idx >> 9, j = idx & 511;
    rv[t][j] = lr[(((long)t * 2 + 1) * 64 + n) * 512 + j];
  }
  __syncthreads();
  const int* arow = adj + ((long)n * 512 + i) * 512;
  float lmax = NEG_INF_F;
  for (int jj = 0; jj < 4; ++jj) {
    int j = tid + jj * 128;
    int a = arow[j];
    float sv;
    if (a > 0) {
      int t = a - 1;
      float xx = lv[t] + rv[t][j];
      sv = xx > 0.f ? xx : LRELU_SLOPE * xx;
    } else {
      sv = NEG_INF_F;
    }
    sc[j] = sv;
    lmax = fmaxf(lmax, sv);
  }
  red[tid] = lmax;
  __syncthreads();
  for (int o = 64; o > 0; o >>= 1) {
    if (tid < o) red[tid] = fmaxf(red[tid], red[tid + o]);
    __syncthreads();
  }
  float m = red[0];
  __syncthreads();
  float lsum = 0.f;
  for (int jj = 0; jj < 4; ++jj) {
    int j = tid + jj * 128;
    float e = __expf(sc[j] - m);
    sc[j] = e;
    lsum += e;
  }
  red[tid] = lsum;
  __syncthreads();
  for (int o = 64; o > 0; o >>= 1) {
    if (tid < o) red[tid] += red[tid + o];
    __syncthreads();
  }
  float inv = 1.f / red[0];
  for (int jj = 0; jj < 4; ++jj) {
    int j = tid + jj * 128;
    coefT[((long)n * 512 + j) * 512 + i] = f2bf(sc[j] * inv);
  }
}

// --------------------------------------------------------------------------- bf16 WMMA GEMM (512x512x512 per batch)
#if __has_builtin(__builtin_amdgcn_global_load_async_to_lds_b128) && \
    __has_builtin(__builtin_amdgcn_s_wait_asynccnt)
#define ATH_ASYNC_LDS 1
#else
#define ATH_ASYNC_LDS 0
#endif

#define GM 512
#define BM 128
#define BN 128
#define BK 32
#define LDPAD 40  // 32 K-elems + 8 pad; 80B row stride keeps 16B alignment

struct Frag32 { uint4 q0, q1; };  // 32 bytes == v16bf

template <bool OUT_BF16>
__global__ __launch_bounds__(128)
void k_gemm_bf16_wmma(const unsigned short* __restrict__ A, long aBatch,
                      const unsigned short* __restrict__ B, long bBatch,
                      void* __restrict__ Cout, long cBatch)
{
  __shared__ unsigned short As[BM][LDPAD];  // M x K, row-major
  __shared__ unsigned short Bt[BN][LDPAD];  // N x K (transposed B tile)
  const int tid = threadIdx.x;              // 128 threads = 4 waves (wave32)
  const int lane = tid & 31;
  const int wid = tid >> 5;
  const int waveM = (wid >> 1) * 64, waveN = (wid & 1) * 64;
  const int nb = blockIdx.z;
  const int blockM = blockIdx.y * BM, blockN = blockIdx.x * BN;
  const unsigned short* Ab = A + (long)nb * aBatch;
  const unsigned short* Bb = B + (long)nb * bBatch;

  v8f acc[4][4];
  const v8f vzero = {0.f, 0.f, 0.f, 0.f, 0.f, 0.f, 0.f, 0.f};
#pragma unroll
  for (int i = 0; i < 4; ++i)
#pragma unroll
    for (int j = 0; j < 4; ++j) acc[i][j] = vzero;

  const int mrow = lane & 15;           // row (A) / col (B) within 16-tile
  const int half = lane >> 4;           // lane half selects K grouping
  const int c0 = half ? 8 : 0;          // A frag K base per ISA layout

  for (int k0 = 0; k0 < GM; k0 += BK) {
    __syncthreads();
    // ---- A tile 128x32 bf16 = 512 x 16B chunks; async copy to LDS if avail
#pragma unroll
    for (int c = 0; c < 4; ++c) {
      int chunk = tid + c * 128;
      int row = chunk >> 2;
      int col = (chunk & 3) * 8;
#if ATH_ASYNC_LDS
      __builtin_amdgcn_global_load_async_to_lds_b128(
          (__attribute__((address_space(1))) v4i*)(Ab + (long)(blockM + row) * GM + k0 + col),
          (__attribute__((address_space(3))) v4i*)&As[row][col], 0, 0);
#else
      *(uint4*)&As[row][col] =
          *(const uint4*)(Ab + (long)(blockM + row) * GM + k0 + col);
#endif
    }
    // ---- B tile 32x128 bf16, stored transposed (Bt[n][k])
#pragma unroll
    for (int c = 0; c < 4; ++c) {
      int chunk = tid + c * 128;
      int k = chunk >> 4;
      int n0 = (chunk & 15) * 8;
      uint4 v = *(const uint4*)(Bb + (long)(k0 + k) * GM + blockN + n0);
      const unsigned short* e = (const unsigned short*)&v;
#pragma unroll
      for (int j = 0; j < 8; ++j) Bt[n0 + j][k] = e[j];
    }
#if ATH_ASYNC_LDS
    __builtin_amdgcn_s_wait_asynccnt(0);
#endif
    __syncthreads();

    // ---- fragments (DS b128 loads) + 16 WMMA per wave per K-step
    Frag32 aq[4], bq[4];
#pragma unroll
    for (int mt = 0; mt < 4; ++mt) {
      const unsigned short* ar = &As[waveM + mt * 16 + mrow][0];
      aq[mt].q0 = *(const uint4*)(ar + c0);
      aq[mt].q1 = *(const uint4*)(ar + c0 + 16);
    }
#pragma unroll
    for (int nt = 0; nt < 4; ++nt) {
      const unsigned short* br = &Bt[waveN + nt * 16 + mrow][0];
      bq[nt].q0 = *(const uint4*)(br + half * 8);
      bq[nt].q1 = *(const uint4*)(br + half * 8 + 16);
    }
#pragma unroll
    for (int mt = 0; mt < 4; ++mt) {
      v16bf av = __builtin_bit_cast(v16bf, aq[mt]);
#pragma unroll
      for (int nt = 0; nt < 4; ++nt) {
        v16bf bv = __builtin_bit_cast(v16bf, bq[nt]);
        acc[mt][nt] = __builtin_amdgcn_wmma_f32_16x16x32_bf16(
            false, av, false, bv, (short)0, acc[mt][nt], false, false);
      }
    }
  }

  // ---- store C (VGPR r -> M = r + half*8 per ISA C/D layout)
#pragma unroll
  for (int mt = 0; mt < 4; ++mt) {
#pragma unroll
    for (int nt = 0; nt < 4; ++nt) {
      int ncol = blockN + waveN + nt * 16 + mrow;
#pragma unroll
      for (int r = 0; r < 8; ++r) {
        int mr = blockM + waveM + mt * 16 + r + half * 8;
        float v = acc[mt][nt][r];
        if constexpr (OUT_BF16) {
          ((unsigned short*)Cout)[(long)nb * cBatch + (long)mr * GM + ncol] = f2bf(v);
        } else {
          ((float*)Cout)[(long)nb * cBatch + (long)mr * GM + ncol] = v;
        }
      }
    }
  }
}

// ---------------------------------------------------------------------------
extern "C" void kernel_launch(void* const* d_in, const int* in_sizes, int n_in,
                              void* d_out, int out_size, void* d_ws, size_t ws_size,
                              hipStream_t stream)
{
  (void)in_sizes; (void)n_in; (void)out_size; (void)ws_size;
  const float* input_state = (const float*)d_in[0];   // [64,512,512]
  const int*   adj         = (const int*)d_in[1];     // [64,512,512]
  /* d_in[2] entity_mask unused by reference */
  const float* query_vec   = (const float*)d_in[3];   // [64,512]
  const float* W_type      = (const float*)d_in[4];   // [3,512,512]
  const float* a_type      = (const float*)d_in[5];   // [3,1024]
  const float* qattn_W1    = (const float*)d_in[6];   // [3,512,1024]
  const float* qattn_W2    = (const float*)d_in[7];   // [3,1024,1024]

  char* ws = (char*)d_ws;
  size_t off = 0;
  auto wsAlloc = [&](size_t bytes) -> void* {
    void* p = ws + off;
    off += (bytes + 255) & ~(size_t)255;
    return p;
  };
  unsigned short* inBF  = (unsigned short*)wsAlloc(64L * 512 * 512 * 2);  // 32 MB
  unsigned short* h2BF  = (unsigned short*)wsAlloc(64L * 512 * 512 * 2);  // 32 MB
  unsigned short* coefT = (unsigned short*)wsAlloc(64L * 512 * 512 * 2);  // 32 MB
  unsigned short* W2BF  = (unsigned short*)wsAlloc(512L * 512 * 2);
  float* gates = (float*)wsAlloc(3L * 64 * 1024 * 4);
  float* u     = (float*)wsAlloc(3L * 2 * 64 * 512 * 4);
  float* lr    = (float*)wsAlloc(3L * 2 * 64 * 512 * 4);

  k_f32_to_bf16<<<2048, 256, 0, stream>>>(input_state, inBF, 16777216L);
  k_f32_to_bf16<<<256, 256, 0, stream>>>(W_type + 2L * 262144, W2BF, 262144L);
  k_gates<<<dim3(64, 3), 256, 0, stream>>>(query_vec, qattn_W1, qattn_W2, gates);
  k_uvec<<<dim3(64, 3), 256, 0, stream>>>(W_type, a_type, gates, u);
  k_leftright<<<dim3(64, 32), 256, 0, stream>>>(input_state, u, lr);
  // h2 = X @ W2  (bf16 out)
  k_gemm_bf16_wmma<true><<<dim3(4, 4, 64), 128, 0, stream>>>(
      inBF, 262144L, W2BF, 0L, h2BF, 262144L);
  k_coefT<<<dim3(512, 64), 128, 0, stream>>>(adj, lr, coefT);
  // out = coefsT @ h2  (f32 out)
  k_gemm_bf16_wmma<false><<<dim3(4, 4, 64), 128, 0, stream>>>(
      coefT, 262144L, h2BF, 262144L, (float*)d_out, 262144L);
}
